// BertSelfAttention_73572789781204
// MI455X (gfx1250) — compile-verified, matrix operand order checked by reading, not run on hardware
//
#include <hip/hip_runtime.h>
#include <hip/hip_bf16.h>
#include <stdint.h>

// ---------------- vector types ----------------
typedef __attribute__((ext_vector_type(16))) _Float16 v16h;
typedef __attribute__((ext_vector_type(8)))  _Float16 v8h;
typedef __attribute__((ext_vector_type(8)))  float    v8f;
typedef __attribute__((ext_vector_type(4)))  float    v4f;

// problem constants
#define NB   8
#define NS   512
#define NHID 1024
#define NH   16
#define ND   64
#define NP   1024          // 2 * ATT_SPAN
#define SC_CONTENT 0.14433756729740643f   // 1/sqrt(3*16)  (content and p2c)
#define SC_C2P     0.018042195912175804f  // 1/sqrt(1024*3)

// ---------------- WMMA helpers ----------------
// A/B fragment for v_wmma_f32_16x16x32_f16, loaded from a row-major f16 tile
// in LDS: lane owns row `row`; K mapping: K = h + (h>=8 ? 8 : 0) + (lane>=16 ? 8 : 0)
// => two contiguous 8-half (16B) loads at kbase+koff and kbase+koff+16.
__device__ __forceinline__ v16h frag_h(const _Float16* base, int stride, int row,
                                       int kbase, int lane) {
  int koff = (lane & 16) ? 8 : 0;
  const _Float16* p = base + row * stride + kbase + koff;
  v8h lo = *(const v8h*)(p);
  v8h hi = *(const v8h*)(p + 16);
  return __builtin_shufflevector(lo, hi, 0,1,2,3,4,5,6,7,8,9,10,11,12,13,14,15);
}

// Same fragment but sourced from an f32 row (convert inline).
__device__ __forceinline__ v16h frag_f32(const float* base, int stride, int row,
                                         int kbase, int lane) {
  int koff = (lane & 16) ? 8 : 0;
  const float* p = base + row * stride + kbase + koff;
  v4f a0 = *(const v4f*)(p);
  v4f a1 = *(const v4f*)(p + 4);
  v4f b0 = *(const v4f*)(p + 16);
  v4f b1 = *(const v4f*)(p + 20);
  v16h f;
#pragma unroll
  for (int t = 0; t < 4; ++t) {
    f[t]      = (_Float16)a0[t];
    f[4 + t]  = (_Float16)a1[t];
    f[8 + t]  = (_Float16)b0[t];
    f[12 + t] = (_Float16)b1[t];
  }
  return f;
}

__device__ __forceinline__ v8f wmma16(v16h a, v16h b, v8f c) {
  return __builtin_amdgcn_wmma_f32_16x16x32_f16(false, a, false, b, (short)0, c,
                                                false, false);
}

// ---------------- async global->LDS staging (CDNA5 ASYNCcnt path) ----------------
__device__ __forceinline__ void wait_async_zero() {
  asm volatile("s_wait_asynccnt 0x0" ::: "memory");
}

// Copy a 64-row x 64-half tile (rows 128B) from global into LDS with row stride
// 144B, using global_load_async_to_lds_b128: 512 16B chunks, 2 per thread.
__device__ __forceinline__ void async_tile64(const _Float16* __restrict__ g,
                                             const char* smem_base,
                                             const _Float16* s, int tid) {
  uint32_t sbase = __builtin_amdgcn_groupstaticsize() +
                   (uint32_t)((const char*)s - smem_base);
#pragma unroll
  for (int it = 0; it < 2; ++it) {
    int e = tid + it * 256;
    int r = e >> 3, c = e & 7;
    uint32_t loff = sbase + r * 144 + c * 16;
    const char* ga = (const char*)g + r * 128 + c * 16;
    asm volatile("global_load_async_to_lds_b128 %0, %1, off"
                 :: "v"(loff), "v"(ga) : "memory");
  }
}

// Same, but rows with r0+r >= rmax are zero-filled (pos band past table end).
__device__ __forceinline__ void async_tile64_guard(const _Float16* __restrict__ g,
                                                   const char* smem_base,
                                                   const _Float16* s, int tid,
                                                   int r0, int rmax) {
  uint32_t sbase = __builtin_amdgcn_groupstaticsize() +
                   (uint32_t)((const char*)s - smem_base);
#pragma unroll
  for (int it = 0; it < 2; ++it) {
    int e = tid + it * 256;
    int r = e >> 3, c = e & 7;
    uint32_t loff = sbase + r * 144 + c * 16;
    if (r0 + r < rmax) {
      const char* ga = (const char*)g + r * 128 + c * 16;
      asm volatile("global_load_async_to_lds_b128 %0, %1, off"
                   :: "v"(loff), "v"(ga) : "memory");
    } else {
      *(v8h*)((char*)s + r * 144 + c * 16) = v8h{};
    }
  }
}

// ---------------- Kernel 1: projection GEMM  Y = X * W^T + bias -> f16 ----------------
// Block: 256 threads = 8 waves, 64x64 output tile. Out layout generic:
// off(m,n) = (m>>smshift)*C0 + (n>>6)*C1 + (m & mask)*64 + (n & 63)
__global__ __launch_bounds__(256) void proj_gemm(
    const float* __restrict__ X, const float* __restrict__ W,
    const float* __restrict__ bias, _Float16* __restrict__ Y,
    int smshift, int C0, int C1) {
  __shared__ _Float16 As[64 * 72];
  __shared__ _Float16 Bs[64 * 72];
  const int tid = threadIdx.x, lane = tid & 31, w = tid >> 5;
  const int l15 = lane & 15, hi = (lane >> 4) & 1;
  const int n0 = blockIdx.x * 64, m0 = blockIdx.y * 64;

  v8f acc[2] = {v8f{}, v8f{}};
  for (int k0 = 0; k0 < NHID; k0 += 64) {
    for (int e = tid; e < 4096; e += 256) {
      int r = e >> 6, c = e & 63;
      As[r * 72 + c] = (_Float16)X[(size_t)(m0 + r) * NHID + k0 + c];
      Bs[r * 72 + c] = (_Float16)W[(size_t)(n0 + r) * NHID + k0 + c];
    }
    __syncthreads();
#pragma unroll
    for (int tt = 0; tt < 2; ++tt) {
      int t16 = w + tt * 8;
      int ti = t16 >> 2, tj = t16 & 3;
#pragma unroll
      for (int ks = 0; ks < 64; ks += 32) {
        v16h a = frag_h(As, 72, ti * 16 + l15, ks, lane);
        v16h b = frag_h(Bs, 72, tj * 16 + l15, ks, lane);
        acc[tt] = wmma16(a, b, acc[tt]);
      }
    }
    __syncthreads();
  }
  const int smmask = (1 << smshift) - 1;
#pragma unroll
  for (int tt = 0; tt < 2; ++tt) {
    int t16 = w + tt * 8;
    int ti = t16 >> 2, tj = t16 & 3;
    int n = n0 + tj * 16 + l15;
    float bv = bias[n];
#pragma unroll
    for (int r = 0; r < 8; ++r) {
      int m = m0 + ti * 16 + r + 8 * hi;
      float val = acc[tt][r] + bv;
      int off = (m >> smshift) * C0 + (n >> 6) * C1 + (m & smmask) * 64 + (n & 63);
      Y[off] = (_Float16)val;
    }
  }
}

// ---------------- Kernel 2: disentangled attention, one (b,h,64-row strip)/block ----------------
__global__ __launch_bounds__(256) void attn_kernel(
    const _Float16* __restrict__ Qh, const _Float16* __restrict__ Kh,
    const _Float16* __restrict__ Vh, const _Float16* __restrict__ Ph,
    const float* __restrict__ mask, float* __restrict__ out) {
  extern __shared__ char smem[];
  float*    score_s = (float*)smem;                         // 64*512*4 = 131072
  _Float16* q_s     = (_Float16*)(smem + 131072);           // 64*72*2  = 9216
  _Float16* t0_s    = (_Float16*)(smem + 131072 + 9216);    // 9216
  _Float16* t1_s    = (_Float16*)(smem + 131072 + 18432);   // 9216
  float*    mask_s  = (float*)(smem + 131072 + 27648);      // 2048
  float*    inv_s   = (float*)(smem + 131072 + 29696);      // 256

  const int tid = threadIdx.x, lane = tid & 31, w = tid >> 5;
  const int l15 = lane & 15, hi = (lane >> 4) & 1;
  const int bx = blockIdx.x;
  const int strip = bx & 7, hh = (bx >> 3) & 15, bb = bx >> 7;
  const int i0 = strip * 64;

  const _Float16* Qg = Qh + ((size_t)(bb * NH + hh) * NS + i0) * ND;
  const _Float16* Kg = Kh + ((size_t)(bb * NH + hh) * NS) * ND;
  const _Float16* Vg = Vh + ((size_t)(bb * NH + hh) * NS) * ND;
  const _Float16* Pg = Ph + (size_t)hh * NP * ND;

  // ---- Phase 0: async-load q strip + mask row ----
  async_tile64(Qg, smem, q_s, tid);
  for (int e = tid; e < NS; e += 256) mask_s[e] = mask[bb * NS + e];
  wait_async_zero();
  __syncthreads();

  // ---- Phase 1: content scores  q . k^T  (writes every score element) ----
  for (int jc = 0; jc < 8; ++jc) {
    async_tile64(Kg + (size_t)jc * 64 * ND, smem, t0_s, tid);
    wait_async_zero();
    __syncthreads();
#pragma unroll
    for (int tt = 0; tt < 2; ++tt) {
      int t16 = w + tt * 8;
      int ti = t16 >> 2, tj = t16 & 3;
      v8f acc = {};
#pragma unroll
      for (int ks = 0; ks < 64; ks += 32) {
        v16h a = frag_h(q_s, 72, ti * 16 + l15, ks, lane);
        v16h b = frag_h(t0_s, 72, tj * 16 + l15, ks, lane);
        acc = wmma16(a, b, acc);
      }
      int col = jc * 64 + tj * 16 + l15;
#pragma unroll
      for (int r = 0; r < 8; ++r) {
        int il = ti * 16 + r + 8 * hi;
        score_s[il * 512 + col] = acc[r] * SC_CONTENT;
      }
    }
    __syncthreads();
  }

  // ---- Phase 2: c2p banded  q . pos^T, scatter-add at j = i - r + 512 ----
  for (int rc = 0; rc < 10; ++rc) {
    int r0 = i0 + rc * 64;
    async_tile64_guard(Pg + (size_t)r0 * ND, smem, t0_s, tid, r0, NP);
    wait_async_zero();
    __syncthreads();
#pragma unroll
    for (int tt = 0; tt < 2; ++tt) {
      int t16 = w + tt * 8;
      int ti = t16 >> 2, tj = t16 & 3;
      v8f acc = {};
#pragma unroll
      for (int ks = 0; ks < 64; ks += 32) {
        v16h a = frag_h(q_s, 72, ti * 16 + l15, ks, lane);
        v16h b = frag_h(t0_s, 72, tj * 16 + l15, ks, lane);
        acc = wmma16(a, b, acc);
      }
      int rloc = tj * 16 + l15;
#pragma unroll
      for (int r = 0; r < 8; ++r) {
        int il = ti * 16 + r + 8 * hi;
        int j = (i0 + il) - (r0 + rloc) + 512;
        if (j >= 0 && j < 512) score_s[il * 512 + j] += acc[r] * SC_C2P;
      }
    }
    __syncthreads();
  }

  // ---- Phase 3: p2c banded  pos . k^T, scatter-add at i = r + j - 512 ----
  for (int jc = 0; jc < 8; ++jc) {
    async_tile64(Kg + (size_t)jc * 64 * ND, smem, t1_s, tid);
    for (int dr = 0; dr < 2; ++dr) {
      int rc = 7 - jc + dr;           // only rc+jc in {7,8} produce in-strip rows
      int r0 = i0 + rc * 64;
      async_tile64_guard(Pg + (size_t)r0 * ND, smem, t0_s, tid, r0, NP);
      wait_async_zero();
      __syncthreads();
#pragma unroll
      for (int tt = 0; tt < 2; ++tt) {
        int t16 = w + tt * 8;
        int ti = t16 >> 2, tj = t16 & 3;
        v8f acc = {};
#pragma unroll
        for (int ks = 0; ks < 64; ks += 32) {
          v16h a = frag_h(t0_s, 72, ti * 16 + l15, ks, lane);
          v16h b = frag_h(t1_s, 72, tj * 16 + l15, ks, lane);
          acc = wmma16(a, b, acc);
        }
        int jp = tj * 16 + l15;
#pragma unroll
        for (int r = 0; r < 8; ++r) {
          int M = ti * 16 + r + 8 * hi;
          int il = (rc + jc) * 64 - 512 + M + jp;
          if (il >= 0 && il < 64)
            score_s[il * 512 + jc * 64 + jp] += acc[r] * SC_CONTENT;
        }
      }
      __syncthreads();
    }
  }

  // ---- Phase 4: softmax per row (mask added; normalization deferred) ----
  for (int rr = 0; rr < 8; ++rr) {
    int row = w * 8 + rr;
    float* srow = score_s + row * 512;
    float mx = -3.0e38f;
    for (int t = 0; t < 16; ++t) {
      int j = lane + t * 32;
      mx = fmaxf(mx, srow[j] + mask_s[j]);
    }
    for (int off = 16; off; off >>= 1) mx = fmaxf(mx, __shfl_xor(mx, off, 32));
    float sum = 0.f;
    for (int t = 0; t < 16; ++t) {
      int j = lane + t * 32;
      float e = __expf(srow[j] + mask_s[j] - mx);
      srow[j] = e;
      sum += e;
    }
    for (int off = 16; off; off >>= 1) sum += __shfl_xor(sum, off, 32);
    if (lane == 0) inv_s[row] = 1.0f / sum;
  }
  __syncthreads();

  // ---- Phase 5: out = exp(score) @ V, scaled by 1/rowsum at the end ----
  v8f oacc[2] = {v8f{}, v8f{}};
  for (int jc = 0; jc < 8; ++jc) {
    // stage V chunk TRANSPOSED: t0_s[d][j'] so B-fragments are contiguous
    // (transpose => cannot use the async DMA path; VALU copy)
    for (int e = tid; e < 4096; e += 256) {
      int jj = e >> 6, d = e & 63;
      t0_s[d * 72 + jj] = Vg[(size_t)(jc * 64 + jj) * ND + d];
    }
    __syncthreads();
#pragma unroll
    for (int tt = 0; tt < 2; ++tt) {
      int t16 = w + tt * 8;
      int ti = t16 >> 2, tj = t16 & 3;
#pragma unroll
      for (int ks = 0; ks < 64; ks += 32) {
        v16h a = frag_f32(score_s, 512, ti * 16 + l15, jc * 64 + ks, lane);
        v16h b = frag_h(t0_s, 72, tj * 16 + l15, ks, lane);
        oacc[tt] = wmma16(a, b, oacc[tt]);
      }
    }
    __syncthreads();
  }
#pragma unroll
  for (int tt = 0; tt < 2; ++tt) {
    int t16 = w + tt * 8;
    int ti = t16 >> 2, tj = t16 & 3;
    int dcol = tj * 16 + l15;
#pragma unroll
    for (int r = 0; r < 8; ++r) {
      int il = ti * 16 + r + 8 * hi;
      float val = oacc[tt][r] * inv_s[il];
      out[((size_t)bb * NS + i0 + il) * NHID + hh * ND + dcol] = val;
    }
  }
}

// ---------------- launcher ----------------
extern "C" void kernel_launch(void* const* d_in, const int* in_sizes, int n_in,
                              void* d_out, int out_size, void* d_ws, size_t ws_size,
                              hipStream_t stream) {
  (void)in_sizes; (void)n_in; (void)out_size; (void)ws_size;
  const float* hidden  = (const float*)d_in[0];
  const float* amask   = (const float*)d_in[1];
  // d_in[2] relative_pos: structurally rel[i,j] = i - j; folded into indexing.
  const float* Wq  = (const float*)d_in[3];
  const float* bq  = (const float*)d_in[4];
  const float* Wk  = (const float*)d_in[5];
  const float* bk  = (const float*)d_in[6];
  const float* Wv  = (const float*)d_in[7];
  const float* bv  = (const float*)d_in[8];
  const float* Wpk = (const float*)d_in[9];
  const float* bpk = (const float*)d_in[10];
  const float* rel = (const float*)d_in[11];
  float* out = (float*)d_out;

  const size_t perMat = (size_t)NB * NH * NS * ND;  // 4,194,304 halves
  _Float16* Qh = (_Float16*)d_ws;
  _Float16* Kh = Qh + perMat;
  _Float16* Vh = Kh + perMat;
  _Float16* Ph = Vh + perMat;                        // NH*NP*ND = 1,048,576 halves

  dim3 blk(256);
  // q/k/v: M=4096 rows, out layout [B,H,S,D]: C0=16*512*64, C1=512*64, smshift=9
  proj_gemm<<<dim3(16, 64), blk, 0, stream>>>(hidden, Wq, bq, Qh, 9, 524288, 32768);
  proj_gemm<<<dim3(16, 64), blk, 0, stream>>>(hidden, Wk, bk, Kh, 9, 524288, 32768);
  proj_gemm<<<dim3(16, 64), blk, 0, stream>>>(hidden, Wv, bv, Vh, 9, 524288, 32768);
  // pos: M=1024 rows, out layout [H,2P,D]: C1=1024*64, smshift=10 (m>>10 == 0)
  proj_gemm<<<dim3(16, 16), blk, 0, stream>>>(rel, Wpk, bpk, Ph, 10, 0, 65536);

  const size_t smem_bytes = 131072 + 3 * 9216 + 2048 + 256;  // 161024
  attn_kernel<<<dim3(NB * NH * (NS / 64)), blk, smem_bytes, stream>>>(
      Qh, Kh, Vh, Ph, amask, out);
}
